// ReverbNetwork_2499670966582
// MI455X (gfx1250) — matrix operand order repaired and checked by reading, not compiled, not run on hardware
//
#include <hip/hip_runtime.h>
#include <stdint.h>
#include <math.h>

// Problem constants (match reference)
#define N_NODES   64
#define N1_NODES  65
#define E_EDGES   513
#define BB        8
#define CC        3
#define HH        64
#define WW        64
#define KK        4
#define HW        (HH * WW)
#define PLANE     (BB * CC * HW)     // elements per node = 98304
#define EQ_VAL    (-0.1f)
#define KTOT      (CC * KK * KK)     // 48, padded to 64 (2 x K32 chunks)
#define TILE_W    80                 // halo col ix=-1 at col 7, interior at col 8 (16B aligned)
#define TILE_ROWS (CC * KK)          // 12
#define TILE_ELEMS (TILE_ROWS * TILE_W) // 960 u16 = 1920 B

typedef __attribute__((ext_vector_type(16))) __bf16         v16bf;
typedef __attribute__((ext_vector_type(16))) unsigned short v16u;
typedef __attribute__((ext_vector_type(8)))  float          v8f;

__device__ __forceinline__ unsigned short f32_to_bf16(float f) {
  union { float f; uint32_t u; } x{f};
  uint32_t u = x.u;
  if ((u & 0x7fffffffu) > 0x7f800000u) return (unsigned short)0x7fc0; // NaN
  u += 0x7fffu + ((u >> 16) & 1u);   // round to nearest even
  return (unsigned short)(u >> 16);
}

// ---------------- Phase 1: acts = sigmoid(states + noise) -> bf16 ----------
__global__ void act_kernel(const float* __restrict__ states,
                           const float* __restrict__ noise,
                           unsigned short* __restrict__ acts, int n) {
  int i = blockIdx.x * blockDim.x + threadIdx.x;
  if (i < n) {
    float v = states[i] + noise[i];
    float s = 1.0f / (1.0f + __expf(-v));
    acts[i] = f32_to_bf16(s);
  }
}

// ---- Phase 2a: pack conv weights into WMMA B-fragment register order ------
// Layout: wB[((e*2 + chunk)*32 + lane)*16 + elem], B 32x16 (KxN):
// lane 0-15 -> col n=lane, K = chunk*32 + 0..15 ; lane 16-31 -> K += 16.
// K index kappa = ci*16 + ky*4 + kx ; kappa>=48 or n>=3 are zero padding.
__global__ void pack_w_kernel(const float* __restrict__ w,
                              unsigned short* __restrict__ wB) {
  int e     = blockIdx.x;
  int chunk = threadIdx.x >> 5;
  int lane  = threadIdx.x & 31;
  int kbase = chunk * 32 + ((lane >= 16) ? 16 : 0);
  int n     = lane & 15;
  unsigned short* dst = wB + ((size_t)(e * 2 + chunk) * 32 + lane) * 16;
  for (int i = 0; i < 16; ++i) {
    int kappa = kbase + i;
    unsigned short v = 0;
    if (kappa < KTOT && n < CC) {
      int ci = kappa >> 4, rem = kappa & 15, ky = rem >> 2, kx = rem & 3;
      v = f32_to_bf16(w[(((size_t)e * CC + n) * CC + ci) * (KK * KK) + ky * KK + kx]);
    }
    dst[i] = v;
  }
}

// ---- Phase 2b: deterministic per-dst edge lists (no atomics) --------------
__global__ void edgelist_kernel(const int* __restrict__ dst,
                                int* __restrict__ counts,
                                int* __restrict__ lists) {
  int n = threadIdx.x;                       // 64 threads, one per dst node
  int c = 0;
  for (int e = 0; e < E_EDGES; ++e)
    if (dst[e] == n) lists[n * E_EDGES + (c++)] = e;
  counts[n] = c;
}

// ---- Phase 3: per-(row, dst-node) implicit-GEMM conv + fused epilogue -----
// Block = 128 threads (4 waves); wave w owns pixels x0=w*16 .. x0+15 of row
// (b,y). Loops over all edges targeting `node`, chaining the WMMA accumulator
// (C->D in place) to realize segment_sum for free. Activation halo strips are
// staged L2->LDS with GLOBAL_LOAD_ASYNC_TO_LDS_B128 (ASYNCcnt path); the
// zero halo is written once per block since it only depends on y.
__global__ __launch_bounds__(128) void conv_kernel(
    const unsigned short* __restrict__ acts,   // bf16 [N1][B][C][H][W]
    const unsigned short* __restrict__ wB,     // packed B fragments
    const float* __restrict__ conv_b,          // [E][C]
    const int*   __restrict__ srcIdx,          // [E]
    const int*   __restrict__ counts,          // [64]
    const int*   __restrict__ lists,           // [64][E]
    const float* __restrict__ states,          // [N1][B][C][H][W]
    const float* __restrict__ resist,          // [N][B][C][H][W]
    float*       __restrict__ out) {
  __shared__ __align__(16) unsigned short tile[TILE_ELEMS]; // 12 rows x 80 cols

  const int node = blockIdx.y;
  const int row  = blockIdx.x;
  const int b    = row / HH;
  const int y    = row % HH;
  const int tid  = threadIdx.x;
  const int lane = tid & 31;
  const int wave = tid >> 5;
  const int x0   = wave * 16;          // 4 waves cover W=64
  const int m    = lane & 15;          // M row (pixel within 16-strip)
  const int hi   = (lane >> 4) & 1;    // lane>=16 half
  const int col  = lane & 15;          // output channel (only 0..2 live)

  // Zero the whole tile once (halo cells stay zero for all edges).
  {
    uint4 z = {0u, 0u, 0u, 0u};
    for (int t = tid; t < TILE_ELEMS / 8; t += 128)
      reinterpret_cast<uint4*>(tile)[t] = z;
  }

  // Per-thread async-copy role: 12 in-plane rows x 8 segments of 16 bytes.
  const int srow = tid >> 3;           // 0..15 (12 live)
  const int sseg = tid & 7;            // 0..7
  const int sci  = srow >> 2;
  const int sky  = srow & 3;
  const int siy  = y - 1 + sky;
  const bool doStage = (tid < 96) && (siy >= 0) && (siy < HH);
  const uint32_t ldsAddr =
      (uint32_t)(uintptr_t)(&tile[srow * TILE_W + 8 + sseg * 8]);
  const size_t gRowOff = (size_t)sci * HW + (size_t)(siy < 0 ? 0 : siy) * WW + sseg * 8;

  v8f   acc = {};
  float biasSum = 0.0f;
  const int  cnt = counts[node];
  const int* lst = lists + node * E_EDGES;

  int prev_s = -1;
  for (int j = 0; j < cnt; ++j) {
    const int e = lst[j];
    const int s = srcIdx[e];
    biasSum += conv_b[e * CC + (col < CC ? col : 0)];  // lanes>=3 never store

    // Prefetch next edge's rows into L2/WGP$ while this edge computes.
    if (j + 1 < cnt) {
      const int s2 = srcIdx[lst[j + 1]];
      if (doStage && sseg == 0)
        __builtin_prefetch(acts + ((size_t)s2 * BB + b) * (CC * HW) + gRowOff, 0, 0);
    }

    if (s != prev_s) {                 // uniform per block: barrier-safe
      prev_s = s;
      __syncthreads();                 // prior edge's LDS reads complete
      const unsigned short* base = acts + ((size_t)s * BB + b) * (CC * HW);
      if (doStage) {
        uint64_t ga = (uint64_t)(uintptr_t)(base + gRowOff);
        asm volatile("global_load_async_to_lds_b128 %0, %1, off"
                     :: "v"(ldsAddr), "v"(ga) : "memory");
      }
      asm volatile("s_wait_asynccnt 0x0" ::: "memory");
      __syncthreads();                 // staged tile visible to all waves
    }

    // Two K32 chunks: kappa = ci*16 + ky*4 + kx (48 live, rest zero)
#pragma unroll
    for (int chunk = 0; chunk < 2; ++chunk) {
      v16u au;
      const int koffb = chunk * 32 + hi * 16;
#pragma unroll
      for (int i = 0; i < 16; ++i) {
        // A layout matches packed B: elems 0..15 -> K koffb+0..15
        int kappa = koffb + i;
        unsigned short v = 0;
        if (kappa < KTOT) {
          int ci = kappa >> 4, rem = kappa & 15, ky = rem >> 2, kx = rem & 3;
          v = tile[(ci * KK + ky) * TILE_W + x0 + m + kx + 7]; // ix=-1 at col 7
        }
        au[i] = v;
      }
      v16u bu = *reinterpret_cast<const v16u*>(
          wB + ((size_t)(e * 2 + chunk) * 32 + lane) * 16);
      v16bf a    = *reinterpret_cast<v16bf*>(&au);
      v16bf bfra = *reinterpret_cast<v16bf*>(&bu);
      acc = __builtin_amdgcn_wmma_f32_16x16x32_bf16(
          false, a, false, bfra, (short)0, acc, false, false);
    }
  }

  // Epilogue: D layout -> lane col (0..2) holds rows r + 8*hi = pixel x-x0
  if (col < CC) {
    size_t pbase = ((((size_t)node * BB + b) * CC + col) * HH + y) * WW;
#pragma unroll
    for (int r = 0; r < 8; ++r) {
      size_t idx = pbase + (size_t)(x0 + hi * 8 + r);
      float sv = states[idx];
      float rv = resist[idx];
      out[idx] = sv + acc[r] + biasSum + fabsf(rv) * (EQ_VAL - sv);
    }
  }
}

// ---- Phase 4: virtual input node N: agg=0, no resistive term --------------
__global__ void copy_node_kernel(const float* __restrict__ states,
                                 float* __restrict__ out, int n) {
  int i = blockIdx.x * blockDim.x + threadIdx.x;
  if (i < n)
    out[(size_t)N_NODES * PLANE + i] = states[(size_t)N_NODES * PLANE + i];
}

extern "C" void kernel_launch(void* const* d_in, const int* in_sizes, int n_in,
                              void* d_out, int out_size, void* d_ws, size_t ws_size,
                              hipStream_t stream) {
  const float* states = (const float*)d_in[0];
  const float* noise  = (const float*)d_in[1];
  const float* conv_w = (const float*)d_in[2];
  const float* conv_b = (const float*)d_in[3];
  const float* resist = (const float*)d_in[4];
  const int*   src    = (const int*)d_in[5];
  const int*   dst    = (const int*)d_in[6];
  float*       out    = (float*)d_out;

  // Workspace carve-up (~13.3 MB total)
  const size_t actsElems = (size_t)N1_NODES * PLANE;          // 6,389,760
  unsigned short* acts = (unsigned short*)d_ws;
  unsigned short* wB   = acts + actsElems;                    // +12,779,520 B (32B aligned)
  int* counts = (int*)(wB + (size_t)E_EDGES * 2 * 32 * 16);   // +1,050,624 B
  int* lists  = counts + 64;                                  // 64*513 ints

  act_kernel<<<(int)((actsElems + 255) / 256), 256, 0, stream>>>(
      states, noise, acts, (int)actsElems);
  pack_w_kernel<<<E_EDGES, 64, 0, stream>>>(conv_w, wB);
  edgelist_kernel<<<1, 64, 0, stream>>>(dst, counts, lists);
  conv_kernel<<<dim3(BB * HH, N_NODES), 128, 0, stream>>>(
      acts, wB, conv_b, src, counts, lists, states, resist, out);
  copy_node_kernel<<<(PLANE + 255) / 256, 256, 0, stream>>>(states, out, PLANE);
}
